// GCNNet_1conv_88553635709223
// MI455X (gfx1250) — matrix operand order repaired and checked by reading, not compiled, not run on hardware
//
#include <hip/hip_runtime.h>
#include <hip/hip_bf16.h>

typedef __attribute__((ext_vector_type(16))) __bf16 v16bf;
typedef __attribute__((ext_vector_type(8)))  __bf16 v8bf;
typedef __attribute__((ext_vector_type(8)))  float  v8f;

#define N_NODES   50000
#define N_EDGES   1600000
#define D_IN      256
#define HIDDEN    1024
#define N_CLASSES 10

// ---------------------------------------------------------------------------
// Degree / normalization
// ---------------------------------------------------------------------------
__global__ void k_deg_init(float* __restrict__ deg) {
    int i = blockIdx.x * 256 + threadIdx.x;
    if (i < N_NODES) deg[i] = 1.0f;   // self-loop weight
}

__global__ void k_deg_accum(const long long* __restrict__ dst,
                            const float* __restrict__ ew,
                            float* __restrict__ deg) {
    int e = blockIdx.x * 256 + threadIdx.x;
    if (e < N_EDGES) {
        unsafeAtomicAdd(&deg[(int)dst[e]], ew[e]);   // global_atomic_add_f32
    }
}

__global__ void k_deg_rsqrt(float* __restrict__ deg) {
    int i = blockIdx.x * 256 + threadIdx.x;
    if (i < N_NODES) {
        float d = deg[i];
        deg[i] = (d > 0.0f) ? rsqrtf(d) : 0.0f;
    }
}

// ---------------------------------------------------------------------------
// fp32 -> bf16 conversion helpers
// ---------------------------------------------------------------------------
__global__ void k_f32_to_bf16(const float* __restrict__ in,
                              __bf16* __restrict__ out, long long n) {
    long long i = (long long)blockIdx.x * 256 + threadIdx.x;
    if (i < n) out[i] = (__bf16)in[i];
}

// in: [rows, cols] f32 row-major  ->  out: [cols, rows] bf16 (B transposed)
__global__ void k_transpose_bf16(const float* __restrict__ in,
                                 __bf16* __restrict__ out,
                                 int rows, int cols) {
    int idx = blockIdx.x * 256 + threadIdx.x;
    if (idx < rows * cols) {
        int r = idx / cols, c = idx % cols;
        out[(size_t)c * rows + r] = (__bf16)in[idx];
    }
}

__global__ void k_relu_cvt(const float* __restrict__ in,
                           __bf16* __restrict__ out, long long n) {
    long long i = (long long)blockIdx.x * 256 + threadIdx.x;
    if (i < n) {
        float v = in[i];
        out[i] = (__bf16)(v > 0.0f ? v : 0.0f);
    }
}

// ---------------------------------------------------------------------------
// Register-blocked WMMA GEMM:
//   C[M,Nout] = A[M,K] (bf16 row-major) * Bt[Nout,K]^T (bf16)
// Block = 256 threads = 8 waves.  Each wave computes a 32x64 output block
// (2 M-tiles x 4 N-tiles of 16x16, 8 fp32 accumulators -> 8 WMMAs per K=32
// step at 1.5 b128 loads per WMMA, A fragments reused 4x, B fragments 2x).
// The 8 waves of a block share the same 32 rows (A hot in L0) and cover 512
// consecutive columns:
//   rows = blockIdx.y*32 .. +31,  cols = blockIdx.x*512 + wave*64 .. +63
// M = N_NODES (last row-block: loads clamped, stores guarded).
// ---------------------------------------------------------------------------
template <int KDIM, bool RELU_BIAS>
__global__ void k_wmma_gemm(const __bf16* __restrict__ A,
                            const __bf16* __restrict__ Bt,
                            const float* __restrict__ bias,
                            float* __restrict__ C, int Nout) {
    const int lane  = threadIdx.x & 31;
    const int wave  = threadIdx.x >> 5;
    const int lmod  = lane & 15;
    const int hi    = lane >> 4;                    // 0 or 1 (lane half)
    const int tileM = blockIdx.y * 32;
    const int tileN = blockIdx.x * 512 + wave * 64;

    // A row pointers (clamped for the ragged final row-block)
    const __bf16* aptr[2];
#pragma unroll
    for (int mi = 0; mi < 2; ++mi) {
        int r = tileM + mi * 16 + lmod;
        if (r > N_NODES - 1) r = N_NODES - 1;
        aptr[mi] = A + (size_t)r * KDIM;
    }
    // B column pointers
    const __bf16* bptr[4];
#pragma unroll
    for (int ni = 0; ni < 4; ++ni)
        bptr[ni] = Bt + (size_t)(tileN + ni * 16 + lmod) * KDIM;

    v8f c[2][4];
#pragma unroll
    for (int mi = 0; mi < 2; ++mi)
#pragma unroll
        for (int ni = 0; ni < 4; ++ni)
            c[mi][ni] = (v8f){};

    for (int k0 = 0; k0 < KDIM; k0 += 32) {
        // Prefetch the A stream a chunk ahead.  Locality 3 -> WGP-scope
        // global_prefetch_b8, which pulls into ALL cache levels (SYS/DEV
        // scope would stop at GL2 and never feed the L0 this loop waits on).
        if (k0 + 128 < KDIM) {
            __builtin_prefetch(aptr[0] + k0 + 128, 0, 3);
            __builtin_prefetch(aptr[1] + k0 + 128, 0, 3);
        }
        // A 16x32 bf16 fragments (interleaved halves per ISA 7.12.2):
        //   elems 0..7  -> K = k0 + hi*8 + t ; elems 8..15 -> +16
        const int kbA = k0 + hi * 8;
        v16bf a[2];
#pragma unroll
        for (int mi = 0; mi < 2; ++mi) {
            v8bf a_lo = *(const v8bf*)(aptr[mi] + kbA);
            v8bf a_hi = *(const v8bf*)(aptr[mi] + kbA + 16);
#pragma unroll
            for (int t = 0; t < 8; ++t) {
                a[mi][t]     = a_lo[t];
                a[mi][t + 8] = a_hi[t];
            }
        }
        // B 32x16 bf16 fragments (same interleaved halves as A):
        //   elems 0..7  -> K = k0 + hi*8 + t ; elems 8..15 -> +16
        const int kbB = k0 + hi * 8;
        v16bf b[4];
#pragma unroll
        for (int ni = 0; ni < 4; ++ni) {
            v8bf b_lo = *(const v8bf*)(bptr[ni] + kbB);
            v8bf b_hi = *(const v8bf*)(bptr[ni] + kbB + 16);
#pragma unroll
            for (int t = 0; t < 8; ++t) {
                b[ni][t]     = b_lo[t];
                b[ni][t + 8] = b_hi[t];
            }
        }
        // 8 WMMAs reusing each A fragment 4x and each B fragment 2x
#pragma unroll
        for (int mi = 0; mi < 2; ++mi)
#pragma unroll
            for (int ni = 0; ni < 4; ++ni)
                c[mi][ni] = __builtin_amdgcn_wmma_f32_16x16x32_bf16(
                    false, a[mi], false, b[ni],
                    (short)0, c[mi][ni], false, false);
    }

    // Epilogue.  C/D layout: lane col = lmod; VGPR r -> row r + hi*8.
    float bv[4];
#pragma unroll
    for (int ni = 0; ni < 4; ++ni)
        bv[ni] = RELU_BIAS ? bias[tileN + ni * 16 + lmod] : 0.0f;

#pragma unroll
    for (int mi = 0; mi < 2; ++mi) {
#pragma unroll
        for (int r = 0; r < 8; ++r) {
            int m = tileM + mi * 16 + r + hi * 8;
            if (m < N_NODES) {
                float* crow = C + (size_t)m * Nout + tileN + lmod;
#pragma unroll
                for (int ni = 0; ni < 4; ++ni) {
                    float v = c[mi][ni][r];
                    if (RELU_BIAS) {
                        v += bv[ni];
                        v = v > 0.0f ? v : 0.0f;
                    }
                    crow[ni * 16] = v;
                }
            }
        }
    }
}

// ---------------------------------------------------------------------------
// Aggregation: agg[i,:] = b1 + dinv[i]^2 * h[i,:]   (self loop)
// ---------------------------------------------------------------------------
__global__ void k_agg_init(const float* __restrict__ h,
                           const float* __restrict__ b1,
                           const float* __restrict__ dinv,
                           float* __restrict__ agg) {
    int row = blockIdx.x;
    int j   = threadIdx.x;                 // 256 threads * float4 = 1024 cols
    float s = dinv[row] * dinv[row];
    const float4* hr = (const float4*)(h + (size_t)row * HIDDEN);
    const float4* bb = (const float4*)b1;
    float4* ar = (float4*)(agg + (size_t)row * HIDDEN);
    float4 hv = hr[j], bv = bb[j], o;
    o.x = bv.x + s * hv.x;
    o.y = bv.y + s * hv.y;
    o.z = bv.z + s * hv.z;
    o.w = bv.w + s * hv.w;
    ar[j] = o;
}

// One block per edge: agg[dst,:] += (dinv[src]*w*dinv[dst]) * h[src,:]
__global__ void k_agg_edges(const long long* __restrict__ srcv,
                            const long long* __restrict__ dstv,
                            const float* __restrict__ ew,
                            const float* __restrict__ dinv,
                            const float* __restrict__ h,
                            float* __restrict__ agg) {
    int e = blockIdx.x;
    int s = (int)srcv[e];
    int d = (int)dstv[e];
    float coef = dinv[s] * ew[e] * dinv[d];
    const float4* hr = (const float4*)(h + (size_t)s * HIDDEN);
    float* ar = agg + (size_t)d * HIDDEN;
    int j = threadIdx.x;                   // 0..255
    float4 hv = hr[j];
    unsafeAtomicAdd(&ar[j * 4 + 0], coef * hv.x);
    unsafeAtomicAdd(&ar[j * 4 + 1], coef * hv.y);
    unsafeAtomicAdd(&ar[j * 4 + 2], coef * hv.z);
    unsafeAtomicAdd(&ar[j * 4 + 3], coef * hv.w);
}

// ---------------------------------------------------------------------------
// Head: logits = h2 @ fcW2 + fcb2, then log_softmax.  One wave per row.
// ---------------------------------------------------------------------------
__global__ void k_head(const float* __restrict__ h2,
                       const float* __restrict__ W2,
                       const float* __restrict__ b2,
                       float* __restrict__ out) {
    int row  = blockIdx.x * 8 + (threadIdx.x >> 5);
    int lane = threadIdx.x & 31;
    if (row >= N_NODES) return;

    float acc[N_CLASSES];
#pragma unroll
    for (int c = 0; c < N_CLASSES; ++c) acc[c] = 0.0f;

    const float* hr = h2 + (size_t)row * HIDDEN;
    for (int k = lane; k < HIDDEN; k += 32) {
        float xv = hr[k];
        const float* wr = W2 + k * N_CLASSES;
#pragma unroll
        for (int c = 0; c < N_CLASSES; ++c) acc[c] += xv * wr[c];
    }
#pragma unroll
    for (int c = 0; c < N_CLASSES; ++c)
        for (int off = 16; off > 0; off >>= 1)
            acc[c] += __shfl_down(acc[c], off, 32);

    if (lane == 0) {
        float z[N_CLASSES];
        float m = -3.0e38f;
#pragma unroll
        for (int c = 0; c < N_CLASSES; ++c) {
            z[c] = acc[c] + b2[c];
            m = z[c] > m ? z[c] : m;
        }
        float ssum = 0.0f;
#pragma unroll
        for (int c = 0; c < N_CLASSES; ++c) ssum += __expf(z[c] - m);
        float lse = __logf(ssum);
#pragma unroll
        for (int c = 0; c < N_CLASSES; ++c)
            out[(size_t)row * N_CLASSES + c] = z[c] - m - lse;
    }
}

// ---------------------------------------------------------------------------
// Launcher
// ---------------------------------------------------------------------------
extern "C" void kernel_launch(void* const* d_in, const int* in_sizes, int n_in,
                              void* d_out, int out_size, void* d_ws, size_t ws_size,
                              hipStream_t stream) {
    const float*     x    = (const float*)d_in[0];
    const long long* ei   = (const long long*)d_in[1];   // [2, E] int64
    const float*     ea   = (const float*)d_in[2];       // [E, 1]
    const float*     W1   = (const float*)d_in[3];       // [256, 1024]
    const float*     b1   = (const float*)d_in[4];
    const float*     fcW1 = (const float*)d_in[5];       // [1024, 1024]
    const float*     fcb1 = (const float*)d_in[6];
    const float*     fcW2 = (const float*)d_in[7];       // [1024, 10]
    const float*     fcb2 = (const float*)d_in[8];
    float* out = (float*)d_out;

    const long long* src = ei;
    const long long* dst = ei + N_EDGES;

    // Workspace carve-up (256B aligned)
    char* ws = (char*)d_ws;
    size_t off = 0;
    auto carve = [&](size_t bytes) -> void* {
        void* p = ws + off;
        off = (off + bytes + 255) & ~(size_t)255;
        return p;
    };
    float*  h      = (float*)carve((size_t)N_NODES * HIDDEN * 4);   // reused as h2
    float*  agg    = (float*)carve((size_t)N_NODES * HIDDEN * 4);
    __bf16* h1bf   = (__bf16*)carve((size_t)N_NODES * HIDDEN * 2);
    __bf16* xbf    = (__bf16*)carve((size_t)N_NODES * D_IN * 2);
    __bf16* W1t    = (__bf16*)carve((size_t)D_IN * HIDDEN * 2);     // [1024][256]
    __bf16* fcW1t  = (__bf16*)carve((size_t)HIDDEN * HIDDEN * 2);   // [1024][1024]
    float*  dinv   = (float*)carve((size_t)N_NODES * 4);
    (void)ws_size; (void)in_sizes; (void)n_in; (void)out_size;

    const int nb_nodes = (N_NODES + 255) / 256;
    const int nb_edges = (N_EDGES + 255) / 256;

    // 1) weighted in-degree (with self loop) -> dinv
    k_deg_init <<<nb_nodes, 256, 0, stream>>>(dinv);
    k_deg_accum<<<nb_edges, 256, 0, stream>>>(dst, ea, dinv);
    k_deg_rsqrt<<<nb_nodes, 256, 0, stream>>>(dinv);

    // 2) bf16 prep: x, W1^T, fcW1^T
    k_f32_to_bf16   <<<(N_NODES * D_IN + 255) / 256, 256, 0, stream>>>(x, xbf,
                        (long long)N_NODES * D_IN);
    k_transpose_bf16<<<(D_IN * HIDDEN + 255) / 256, 256, 0, stream>>>(W1, W1t,
                        D_IN, HIDDEN);
    k_transpose_bf16<<<(HIDDEN * HIDDEN + 255) / 256, 256, 0, stream>>>(fcW1, fcW1t,
                        HIDDEN, HIDDEN);

    // GEMM grid: 512 cols per block, 32 rows per block (last block ragged)
    const dim3 gemm_grid(HIDDEN / 512, (N_NODES + 31) / 32);

    // 3) h = x @ W1   (WMMA bf16, fp32 accum)
    k_wmma_gemm<D_IN, false><<<gemm_grid, 256, 0, stream>>>(
        xbf, W1t, nullptr, h, HIDDEN);

    // 4) aggregation: agg = b1 + dinv^2*h (self loop) + edge scatter
    k_agg_init <<<N_NODES, 256, 0, stream>>>(h, b1, dinv, agg);
    k_agg_edges<<<N_EDGES, 256, 0, stream>>>(src, dst, ea, dinv, h, agg);

    // 5) h1 = relu(agg) -> bf16
    k_relu_cvt<<<((long long)N_NODES * HIDDEN + 255) / 256, 256, 0, stream>>>(
        agg, h1bf, (long long)N_NODES * HIDDEN);

    // 6) h2 = relu(h1 @ fcW1 + fcb1)  (WMMA bf16; overwrite h)
    k_wmma_gemm<HIDDEN, true><<<gemm_grid, 256, 0, stream>>>(
        h1bf, fcW1t, fcb1, h, HIDDEN);

    // 7) logits + log_softmax
    k_head<<<(N_NODES + 7) / 8, 256, 0, stream>>>(h, fcW2, fcb2, out);
}
